// ScannedRNN_30897994727649
// MI455X (gfx1250) — matrix operand order repaired and checked by reading, not compiled, not run on hardware
//
#include <hip/hip_runtime.h>

#define T_STEPS 512
#define BATCH   128
#define DH      1024
#define GH      3072
#define NWG_SCAN 128

typedef __attribute__((ext_vector_type(16))) __bf16 v16bf;
typedef __attribute__((ext_vector_type(8)))  __bf16 v8bf;
typedef __attribute__((ext_vector_type(8)))  float  v8f;
typedef __attribute__((ext_vector_type(8)))  float  v8fv;
typedef __attribute__((ext_vector_type(4)))  unsigned int v4u;

struct Frag2 { v4u lo, hi; };

static __device__ inline v16bf make_frag(v4u lo, v4u hi) {
  Frag2 f{lo, hi};
  return __builtin_bit_cast(v16bf, f);
}

// Native packed f32 -> bf16 converts (RNE); clang lowers to v_cvt*bf16*.
static __device__ inline unsigned short f2bf(float f) {
  __bf16 h = (__bf16)f;
  return __builtin_bit_cast(unsigned short, h);
}
static __device__ inline float bf2f(unsigned short h) {
  unsigned int u = ((unsigned int)h) << 16;
  return __builtin_bit_cast(float, u);
}
// 8 floats -> 8 bf16 (16 bytes) in one shot.
static __device__ inline v4u cvt8(float4 a, float4 b) {
  v8fv f;
  f[0] = a.x; f[1] = a.y; f[2] = a.z; f[3] = a.w;
  f[4] = b.x; f[5] = b.y; f[6] = b.z; f[7] = b.w;
  v8bf h = __builtin_convertvector(f, v8bf);
  return __builtin_bit_cast(v4u, h);
}

// Flat pointers to LDS carry the LDS byte offset in the low 32 bits
// (aperture id lives in the high bits) — that is exactly what the async
// VDST operand wants.
static __device__ inline unsigned lds_off(const void* p) {
  return (unsigned)(unsigned long long)p;
}
// Async global->LDS copy of 16 bytes for this lane, tracked by ASYNCcnt.
static __device__ inline void async_copy_b128(unsigned lds_byte_off, const void* gsrc) {
  asm volatile("global_load_async_to_lds_b128 %0, %1, off"
               :: "v"(lds_byte_off), "v"(gsrc) : "memory");
}
static __device__ inline void wait_async0() {
  asm volatile("s_wait_asynccnt 0x0" ::: "memory");
}

// ---------------------------------------------------------------------------
// Weight prep: dst[g][k] = bf16(src[k][g]);  src is [DH][GH] f32.
// ---------------------------------------------------------------------------
__global__ void __launch_bounds__(256)
wtrans_bf16(const float* __restrict__ src, unsigned short* __restrict__ dst) {
  __shared__ float tile[32][33];
  const int c0 = blockIdx.x * 32;
  const int r0 = blockIdx.y * 32;
  const int tx = threadIdx.x, ty = threadIdx.y;
#pragma unroll
  for (int j = 0; j < 4; ++j)
    tile[ty + j * 8][tx] = src[(size_t)(r0 + ty + j * 8) * GH + (c0 + tx)];
  __syncthreads();
#pragma unroll
  for (int j = 0; j < 4; ++j)
    dst[(size_t)(c0 + ty + j * 8) * DH + (r0 + tx)] = f2bf(tile[tx][ty + j * 8]);
}

// ---------------------------------------------------------------------------
// Phase 1: xi[m][g] = ins[m][:] @ Wi[:, g] + bi[g] for m in chunk rows.
// BLK 64x64x32, 256 threads = 8 waves (4 M-tiles x 2 N-halves), bf16 WMMA.
// Double-buffered K loop; B tile staged with async global->LDS copies.
// ---------------------------------------------------------------------------
__global__ void __launch_bounds__(256)
gemm_xi(const float* __restrict__ ins,          // [T*B][DH] f32
        const unsigned short* __restrict__ WiT, // [GH][DH] bf16
        const float* __restrict__ bi,           // [GH]
        float* __restrict__ xi,                 // [tc*B][GH] f32 (chunk-local)
        int t0) {
  __shared__ unsigned short At[2][64 * 32];
  __shared__ unsigned short Bt[2][64 * 32];
  const int tid  = threadIdx.x;
  const int lane = tid & 31;
  const int wave = tid >> 5;
  const int wm = wave & 3, wn = wave >> 2;
  const int ln = lane & 15;
  const int hi = lane >> 4;
  const size_t mbase = (size_t)t0 * BATCH + (size_t)blockIdx.x * 64;
  const int g0 = blockIdx.y * 64;

  v8f c0 = {}, c1 = {};
  const int lrow = tid >> 2;       // 0..63
  const int lseg = (tid & 3) * 8;  // 0,8,16,24 (elements)

  const float* aRow = ins + (mbase + lrow) * DH + lseg;
  const unsigned short* bRow = WiT + (size_t)(g0 + lrow) * DH + lseg;

  auto stage = [&](int k0, int b) {
    // B tile: async global -> LDS (already bf16, K-contiguous rows of WiT)
    async_copy_b128(lds_off(&Bt[b][lrow * 32 + lseg]), bRow + k0);
    // A tile: f32 -> bf16 via packed converts
    float4 a0 = *(const float4*)(aRow + k0);
    float4 a1 = *(const float4*)(aRow + k0 + 4);
    *(v4u*)&At[b][lrow * 32 + lseg] = cvt8(a0, a1);
  };

  stage(0, 0);
  const int NK = DH / 32;
  for (int kb = 0; kb < NK; ++kb) {
    const int buf = kb & 1;
    wait_async0();
    __syncthreads();                 // tile[buf] ready; buf^1 free for staging
    if (kb + 1 < NK) stage((kb + 1) * 32, buf ^ 1);

    const int koffA = hi * 16;
    const unsigned short* arow = &At[buf][(wm * 16 + ln) * 32 + koffA];
    v16bf af = make_frag(*(const v4u*)arow, *(const v4u*)(arow + 8));
    const int koffB = hi * 16;
    const v4u* b0p = (const v4u*)&Bt[buf][(wn * 32 + ln) * 32 + koffB];
    const v4u* b1p = (const v4u*)&Bt[buf][(wn * 32 + 16 + ln) * 32 + koffB];
    v16bf bf0 = make_frag(b0p[0], b0p[1]);
    v16bf bf1 = make_frag(b1p[0], b1p[1]);
    c0 = __builtin_amdgcn_wmma_f32_16x16x32_bf16(false, af, false, bf0, (short)0, c0, false, false);
    c1 = __builtin_amdgcn_wmma_f32_16x16x32_bf16(false, af, false, bf1, (short)0, c1, false, false);
  }

  // Epilogue: C layout — reg i, lanes0-15: M = i (+8 for lanes16-31), N = lane&15
  const size_t mrow0 = (size_t)blockIdx.x * 64 + wm * 16 + hi * 8;
  const int gA = g0 + wn * 32 + ln;
  const int gB = gA + 16;
  const float biA = bi[gA], biB = bi[gB];
#pragma unroll
  for (int i = 0; i < 8; ++i) {
    xi[(mrow0 + i) * GH + gA] = c0[i] + biA;
    xi[(mrow0 + i) * GH + gB] = c1[i] + biB;
  }
}

// ---------------------------------------------------------------------------
// Phase 2: persistent scan. Grid = 128 WGs (8 batch-tiles x 16 hidden slices),
// 256 threads = 8 waves: 4 column groups x 2 K-halves (reduced via LDS).
// Per step: async-copy the bf16 h mirror into LDS, zero reset rows,
// hh = h @ Wh (bf16 WMMA, Wh streamed from L2), gates, write h_new (f32 to
// d_out, bf16 to the mirror), grid-wide barrier.
// ---------------------------------------------------------------------------
__global__ void __launch_bounds__(256)
scan_kernel(const float* __restrict__ xi,            // [tc][B][GH]
            const unsigned char* __restrict__ resets,// [T][B]
            const unsigned short* __restrict__ WhT,  // [GH][DH] bf16
            const float* __restrict__ bhn,           // [DH]
            float* __restrict__ out,                 // [T][B][DH]
            unsigned short* __restrict__ hbf,        // [B][DH] bf16 mirror
            unsigned int* __restrict__ cnt,
            int t0, int tc) {
  __shared__ unsigned short hA[16 * DH];   // 32KB: bf16 h_{t-1} (reset applied)
  __shared__ float partial[3 * 4 * 256];   // 12KB: K-half reduction

  const int tid  = threadIdx.x;
  const int lane = tid & 31;
  const int wave = tid >> 5;
  const int w4 = wave & 3;   // column sub-tile
  const int kh = wave >> 2;  // K half
  const int ln = lane & 15;
  const int hi = lane >> 4;
  const int mt = blockIdx.x & 7;
  const int ns = blockIdx.x >> 3;
  const int b0 = mt * 16;
  const int j0 = ns * 64 + w4 * 16;  // hidden column base for this wave
  const int g0r = j0, g0z = DH + j0, g0n = 2 * DH + j0;

  // Per-thread segment for the hA rebuild: row = tid>>4, 64 bf16 per thread.
  const int hrow_i = tid >> 4;
  const int hcs    = (tid & 15) * 64;
  const unsigned hA_dst = lds_off(&hA[hrow_i * DH + hcs]);
  const unsigned short* hsrc = hbf + (size_t)(b0 + hrow_i) * DH + hcs;

  for (int t = 0; t < tc; ++t) {
    const int tg = t0 + t;

    // ---- rebuild hA from the bf16 mirror (async), then zero reset rows ----
    if (tg > 0) {
#pragma unroll
      for (int q = 0; q < 8; ++q)
        async_copy_b128(hA_dst + q * 16, hsrc + q * 8);
    }
    wait_async0();
    {
      const unsigned char rst = resets[(size_t)tg * BATCH + b0 + hrow_i];
      if (tg == 0 || rst) {
        v4u z = {};
        v4u* dstp = (v4u*)&hA[hrow_i * DH + hcs];
#pragma unroll
        for (int q = 0; q < 8; ++q) dstp[q] = z;
      }
    }
    __syncthreads();

    // ---- hh = hA x Wh slice (this wave's K half) ----
    v8f cr = {}, cz = {}, cn = {};
    {
      const int kbase = kh * (DH / 2);
      const unsigned short* hrow = &hA[ln * DH];
      const unsigned short* wr = WhT + (size_t)(g0r + ln) * DH;
      const unsigned short* wz = WhT + (size_t)(g0z + ln) * DH;
      const unsigned short* wn_ = WhT + (size_t)(g0n + ln) * DH;
      const int koffA = hi * 16;
      const int koffB = hi * 16;
      for (int kb = 0; kb < DH / 2; kb += 32) {
        const int k0 = kbase + kb;
        v16bf af = make_frag(*(const v4u*)&hrow[k0 + koffA],
                             *(const v4u*)&hrow[k0 + koffA + 8]);
        const v4u* pr = (const v4u*)&wr[k0 + koffB];
        const v4u* pz = (const v4u*)&wz[k0 + koffB];
        const v4u* pn = (const v4u*)&wn_[k0 + koffB];
        v16bf br = make_frag(pr[0], pr[1]);
        v16bf bz = make_frag(pz[0], pz[1]);
        v16bf bn = make_frag(pn[0], pn[1]);
        cr = __builtin_amdgcn_wmma_f32_16x16x32_bf16(false, af, false, br, (short)0, cr, false, false);
        cz = __builtin_amdgcn_wmma_f32_16x16x32_bf16(false, af, false, bz, (short)0, cz, false, false);
        cn = __builtin_amdgcn_wmma_f32_16x16x32_bf16(false, af, false, bn, (short)0, cn, false, false);
      }
    }

    // ---- reduce K halves ----
    if (kh == 1) {
      float* pp = &partial[w4 * 256 + lane * 8];
#pragma unroll
      for (int i = 0; i < 8; ++i) {
        pp[i] = cr[i]; pp[1024 + i] = cz[i]; pp[2048 + i] = cn[i];
      }
    }
    __syncthreads();

    // ---- gates + state update (kh==0 waves) ----
    if (kh == 0) {
      const float* pp = &partial[w4 * 256 + lane * 8];
      const int j = j0 + ln;
      const float bn_j = bhn[j];
#pragma unroll
      for (int i = 0; i < 8; ++i) {
        const int row = hi * 8 + i;
        const size_t xb = ((size_t)t * BATCH + b0 + row) * GH;
        const float hhr = cr[i] + pp[i];
        const float hhz = cz[i] + pp[1024 + i];
        const float hhn = cn[i] + pp[2048 + i];
        const float xr = xi[xb + j];
        const float xz = xi[xb + DH + j];
        const float xn = xi[xb + 2 * DH + j];
        const float r = 1.f / (1.f + __expf(-(xr + hhr)));
        const float z = 1.f / (1.f + __expf(-(xz + hhz)));
        const float n = tanhf(xn + r * (hhn + bn_j));
        const float hp = bf2f(hA[row * DH + j]);
        const float hnew = (1.f - z) * n + z * hp;
        out[((size_t)tg * BATCH + b0 + row) * DH + j] = hnew;
        hbf[(size_t)(b0 + row) * DH + j] = f2bf(hnew);
      }
    }

    // ---- device-wide step barrier ----
    __threadfence();
    __syncthreads();
    if (tid == 0) {
      __hip_atomic_fetch_add(cnt, 1u, __ATOMIC_RELEASE, __HIP_MEMORY_SCOPE_AGENT);
      const unsigned int target = (unsigned int)(tg + 1) * NWG_SCAN;
      while (__hip_atomic_load(cnt, __ATOMIC_ACQUIRE, __HIP_MEMORY_SCOPE_AGENT) < target) {
        __builtin_amdgcn_s_sleep(2);
      }
    }
    __syncthreads();
    __threadfence();
  }
}

// ---------------------------------------------------------------------------
// Host launcher
// ---------------------------------------------------------------------------
extern "C" void kernel_launch(void* const* d_in, const int* in_sizes, int n_in,
                              void* d_out, int out_size, void* d_ws, size_t ws_size,
                              hipStream_t stream) {
  (void)in_sizes; (void)n_in; (void)out_size;
  const float*         ins    = (const float*)d_in[0];
  const unsigned char* resets = (const unsigned char*)d_in[1];
  const float*         Wi     = (const float*)d_in[2];
  const float*         bi     = (const float*)d_in[3];
  const float*         Wh     = (const float*)d_in[4];
  const float*         bhn    = (const float*)d_in[5];
  float*               out    = (float*)d_out;
  char*                ws     = (char*)d_ws;

  const size_t MB = 1024ull * 1024ull;
  unsigned short* WiT = (unsigned short*)(ws);
  unsigned short* WhT = (unsigned short*)(ws + 6 * MB);
  unsigned int*   cnt = (unsigned int*)(ws + 12 * MB);
  unsigned short* hbf = (unsigned short*)(ws + 12 * MB + 65536);  // 256KB mirror
  float*          xiB = (float*)(ws + 12 * MB + 65536 + 262144);
  const size_t fixed  = 12 * MB + 65536 + 262144;
  const size_t per_t  = (size_t)BATCH * GH * sizeof(float);  // 1.5 MB / step

  // Chunk length limited by scratch capacity.
  long cap = (ws_size > fixed) ? (long)((ws_size - fixed) / per_t) : 1;
  int Tc = (cap < 1) ? 1 : (cap > T_STEPS ? T_STEPS : (int)cap);

  hipMemsetAsync(cnt, 0, 64, stream);

  // One-time weight transpose + bf16 convert (Wi and Wh -> [G][K] bf16).
  dim3 tb(32, 8), tg(GH / 32, DH / 32);
  wtrans_bf16<<<tg, tb, 0, stream>>>(Wi, WiT);
  wtrans_bf16<<<tg, tb, 0, stream>>>(Wh, WhT);

  for (int t0 = 0; t0 < T_STEPS; t0 += Tc) {
    const int tc = (T_STEPS - t0 < Tc) ? (T_STEPS - t0) : Tc;
    dim3 gg((unsigned)(tc * BATCH / 64), GH / 64);
    gemm_xi<<<gg, 256, 0, stream>>>(ins, WiT, bi, xiB, t0);
    scan_kernel<<<NWG_SCAN, 256, 0, stream>>>(xiB, resets, WhT, bhn, out, hbf, cnt, t0, tc);
  }
}
